// WindowAttention_47785806135818
// MI455X (gfx1250) — compile-verified, matrix operand order checked by reading, not run on hardware
//
#include <hip/hip_runtime.h>
#include <hip/hip_bf16.h>
#include <math.h>

// ---------------- WMMA plumbing (CDNA5 / gfx1250, wave32) ----------------
typedef __attribute__((ext_vector_type(16))) _Float16 v16h;
typedef __attribute__((ext_vector_type(8)))  float    v8f;

// 16-bit A/B fragment: element i of the per-lane v16h maps to K index:
//   k = (i&7) + 16*(i>=8) + 8*(lane>=16);  row/col = lane&15
__device__ __forceinline__ int frag_k(int lane, int i) {
    return (i & 7) + ((i >> 3) << 4) + ((lane >> 4) << 3);
}

__device__ __forceinline__ v8f wmma_f16(v16h a, v16h b, v8f c) {
    return __builtin_amdgcn_wmma_f32_16x16x32_f16(
        /*neg_a=*/false, a, /*neg_b=*/false, b,
        /*c_mod=*/(short)0, c, /*reuse_a=*/false, /*reuse_b=*/false);
}

// A fragment (16 x 32) from row-major f16, leading dim ld
__device__ __forceinline__ v16h load_a_f16(const _Float16* p, int ld) {
    int lane = (int)(threadIdx.x & 31);
    int m = lane & 15;
    v16h f;
#pragma unroll
    for (int i = 0; i < 16; ++i) f[i] = p[m * ld + frag_k(lane, i)];
    return f;
}

// A fragment (16 x 32) from row-major f32 (converted on the fly)
__device__ __forceinline__ v16h load_a_f32(const float* p, int ld) {
    int lane = (int)(threadIdx.x & 31);
    int m = lane & 15;
    v16h f;
#pragma unroll
    for (int i = 0; i < 16; ++i) f[i] = (_Float16)p[m * ld + frag_k(lane, i)];
    return f;
}

// B fragment (32 x 16) from K-major storage (row = k), leading dim ld
__device__ __forceinline__ v16h load_b_kmajor(const _Float16* p, int ld) {
    int lane = (int)(threadIdx.x & 31);
    int n = lane & 15;
    v16h f;
#pragma unroll
    for (int i = 0; i < 16; ++i) f[i] = p[frag_k(lane, i) * ld + n];
    return f;
}

// B fragment (32 x 16) where memory is N-major (weight W[o][c]): B[k][n] = W[n][k]
__device__ __forceinline__ v16h load_b_nmajor(const _Float16* p, int ld) {
    int lane = (int)(threadIdx.x & 31);
    int n = lane & 15;
    v16h f;
#pragma unroll
    for (int i = 0; i < 16; ++i) f[i] = p[n * ld + frag_k(lane, i)];
    return f;
}

// ---------------- Problem constants ----------------
#define BWIN   1024      // windows (B_)
#define NTOK   128       // tokens per window
#define CDIM   180
#define KPAD   192       // C padded to multiple of 32
#define HEADS  6
#define HD     30
#define HDP    32        // head dim padded
#define OQKV   540       // 3*C
#define OQKVP  576       // 3*C padded to 36 tiles of 16 (no guards in K loop)
#define C2PAD  384       // 2*C (=360) padded
#define NWIN   64        // mask windows

// ---------------- Prep kernels ----------------
__global__ void prep_x_kernel(const float* __restrict__ x,
                              _Float16* __restrict__ xf,
                              _Float16* __restrict__ xmf) {
    long idx = (long)blockIdx.x * 256 + threadIdx.x;       // over 131072*192
    if (idx >= (long)BWIN * NTOK * KPAD) return;
    int c = (int)(idx % KPAD);
    long r = idx / KPAD;                                   // global token row
    int n = (int)(r % NTOK);
    float v = 0.0f;
    if (c < CDIM) v = x[r * CDIM + c];
    xf[idx] = (_Float16)v;
    float pb = 0.0f;
    if (c < CDIM) {
        int p  = n & 63;                                   // pos bias tiled over 2 halves
        int py = p >> 3, px = p & 7;
        const float two_pi = 6.283185307179586f;
        float emb = (c < 90) ? ((float)(py + 1) * two_pi / (8.0f + 1e-6f))
                             : ((float)(px + 1) * two_pi / (8.0f + 1e-6f));
        int j = (c < 90) ? c : c - 90;
        float t = powf(10000.0f, (2.0f * (float)(j >> 1)) / 90.0f);
        float a = emb / t;
        pb = (j & 1) ? cosf(a) : sinf(a);
    }
    xmf[idx] = (_Float16)(v + pb);
}

__global__ void prep_wqkv_kernel(const float* __restrict__ W, _Float16* __restrict__ Wf) {
    int idx = blockIdx.x * 256 + threadIdx.x;              // 576*192
    if (idx >= OQKVP * KPAD) return;
    int c = idx % KPAD, o = idx / KPAD;
    float v = (o < OQKV && c < CDIM) ? W[o * CDIM + c] : 0.0f;
    Wf[idx] = (_Float16)v;
}

__global__ void prep_wproj_kernel(const float* __restrict__ W, _Float16* __restrict__ Wf) {
    int idx = blockIdx.x * 256 + threadIdx.x;              // 192*384
    if (idx >= 192 * C2PAD) return;
    int c = idx % C2PAD, o = idx / C2PAD;
    float v = (o < CDIM && c < 360) ? W[o * 360 + c] : 0.0f;
    Wf[idx] = (_Float16)v;
}

__global__ void prep_bias_kernel(const float* __restrict__ rpb, float* __restrict__ bias) {
    int idx = blockIdx.x * 256 + threadIdx.x;              // 6*128*128
    if (idx >= HEADS * NTOK * NTOK) return;
    int j = idx % NTOK;
    int i = (idx / NTOK) % NTOK;
    int h = idx / (NTOK * NTOK);
    int di = i >> 6, hi = (i >> 3) & 7, wi = i & 7;
    int dj = j >> 6, hj = (j >> 3) & 7, wj = j & 7;
    int rel = (di - dj + 1) * 225 + (hi - hj + 7) * 15 + (wi - wj + 7);
    bias[idx] = rpb[rel * HEADS + h];
}

// zero the d=30,31 pad lanes of q/v and the k-transposed pad rows
__global__ void pad_qkv_kernel(_Float16* __restrict__ q,
                               _Float16* __restrict__ kT,
                               _Float16* __restrict__ v) {
    int idx = blockIdx.x * 256 + threadIdx.x;              // 1024*6*128
    if (idx >= BWIN * HEADS * NTOK) return;
    int m  = idx % NTOK;
    long bh = idx / NTOK;
    q[(bh * NTOK + m) * HDP + 30] = (_Float16)0.0f;
    q[(bh * NTOK + m) * HDP + 31] = (_Float16)0.0f;
    v[(bh * NTOK + m) * HDP + 30] = (_Float16)0.0f;
    v[(bh * NTOK + m) * HDP + 31] = (_Float16)0.0f;
    kT[(bh * HDP + 30) * NTOK + m] = (_Float16)0.0f;
    kT[(bh * HDP + 31) * NTOK + m] = (_Float16)0.0f;
}

// ---------------- QKV GEMM: Y[tok,o] = sum_c X[tok,c]*W[o,c] ----------------
// grid (BWIN, 9); 256 threads = 8 waves -> 8 row tiles of 16.
// Each wave: 1 row tile x 4 col tiles. Weight rows padded to 576 so the
// K loop is guard-free (no accumulator phi copies).
__device__ __forceinline__ void qkv_scatter(v8f acc, int jt, int row0, int lane, int b,
                                            _Float16* __restrict__ q,
                                            _Float16* __restrict__ kT,
                                            _Float16* __restrict__ v, float qscale) {
#pragma unroll
    for (int r = 0; r < 8; ++r) {
        int m = r + ((lane >> 4) << 3);
        int n = lane & 15;
        int o = jt * 16 + n;
        if (o >= OQKV) continue;
        int s = o / CDIM, rem = o % CDIM;
        int h = rem / HD, d = rem % HD;
        int tok = row0 + m;
        long bh = (long)b * HEADS + h;
        float val = acc[r];
        if (s == 0)      q [(bh * NTOK + tok) * HDP + d] = (_Float16)(val * qscale);
        else if (s == 1) kT[(bh * HDP + d) * NTOK + tok] = (_Float16)val;
        else             v [(bh * NTOK + tok) * HDP + d] = (_Float16)val;
    }
}

__global__ void __launch_bounds__(256)
qkv_gemm_kernel(const _Float16* __restrict__ X, const _Float16* __restrict__ W,
                _Float16* __restrict__ q, _Float16* __restrict__ kT,
                _Float16* __restrict__ v, float qscale) {
    int b    = blockIdx.x;
    int jg   = blockIdx.y;                 // col-tile group: tiles jg*4 .. jg*4+3 (of 36)
    int wave = (int)(threadIdx.x >> 5);
    int lane = (int)(threadIdx.x & 31);
    int row0 = wave * 16;
    const _Float16* A  = X + ((long)b * NTOK + row0) * KPAD;
    const _Float16* Bw = W + (long)jg * 4 * 16 * KPAD;
    v8f acc0 = {}, acc1 = {}, acc2 = {}, acc3 = {};
#pragma unroll
    for (int kc = 0; kc < KPAD / 32; ++kc) {
        v16h a = load_a_f16(A + kc * 32, KPAD);
        if (kc + 1 < KPAD / 32)
            __builtin_prefetch(A + (kc + 1) * 32, 0, 3);   // global_prefetch_b8
        v16h b0 = load_b_nmajor(Bw + 0 * 16 * KPAD + kc * 32, KPAD);
        v16h b1 = load_b_nmajor(Bw + 1 * 16 * KPAD + kc * 32, KPAD);
        v16h b2 = load_b_nmajor(Bw + 2 * 16 * KPAD + kc * 32, KPAD);
        v16h b3 = load_b_nmajor(Bw + 3 * 16 * KPAD + kc * 32, KPAD);
        acc0 = wmma_f16(a, b0, acc0);
        acc1 = wmma_f16(a, b1, acc1);
        acc2 = wmma_f16(a, b2, acc2);
        acc3 = wmma_f16(a, b3, acc3);
    }
    qkv_scatter(acc0, jg * 4 + 0, row0, lane, b, q, kT, v, qscale);
    qkv_scatter(acc1, jg * 4 + 1, row0, lane, b, q, kT, v, qscale);
    qkv_scatter(acc2, jg * 4 + 2, row0, lane, b, q, kT, v, qscale);
    qkv_scatter(acc3, jg * 4 + 3, row0, lane, b, q, kT, v, qscale);
}

// ---------------- Self attention: one block per (window, head) ----------------
__global__ void __launch_bounds__(256)
attn_self_kernel(const _Float16* __restrict__ q, const _Float16* __restrict__ kT,
                 const _Float16* __restrict__ v, const float* __restrict__ bias,
                 const float* __restrict__ mask, _Float16* __restrict__ xout) {
    __shared__ float S[NTOK * NTOK];        // 64 KB
    int b = blockIdx.x / HEADS, h = blockIdx.x % HEADS;
    long bh = (long)b * HEADS + h;
    int wave = (int)(threadIdx.x >> 5), lane = (int)(threadIdx.x & 31);
    const _Float16* qb = q  + bh * NTOK * HDP;
    const _Float16* kb = kT + bh * HDP * NTOK;
    const _Float16* vb = v  + bh * NTOK * HDP;
    const float* biash = bias + (long)h * NTOK * NTOK;
    const float* maskw = mask + (long)(b & (NWIN - 1)) * NTOK * NTOK;

    // S = q*k^T + bias + mask : wave owns column tile j = wave,
    // B (k^T) fragment is loop-invariant -> loaded once, 8 row tiles of A.
    {
        int j = wave;
        v16h bb = load_b_kmajor(kb + j * 16, NTOK);
#pragma unroll
        for (int i = 0; i < 8; ++i) {
            v16h a = load_a_f16(qb + i * 16 * HDP, HDP);
            v8f acc = {};
            acc = wmma_f16(a, bb, acc);
#pragma unroll
            for (int r = 0; r < 8; ++r) {
                int m = i * 16 + r + ((lane >> 4) << 3);
                int n = j * 16 + (lane & 15);
                S[m * NTOK + n] = acc[r] + biash[m * NTOK + n] + maskw[m * NTOK + n];
            }
        }
    }
    __syncthreads();

    // row softmax: 2 lanes per row (64 elems each), combine via shfl_xor(1)
    {
        int row  = (int)(threadIdx.x >> 1);
        int half = (int)(threadIdx.x & 1);
        float* Sr = S + row * NTOK + half * 64;
        float mx = -1e30f;
        for (int j = 0; j < 64; ++j) mx = fmaxf(mx, Sr[j]);
        mx = fmaxf(mx, __shfl_xor(mx, 1));
        float sum = 0.0f;
        for (int j = 0; j < 64; ++j) { float e = __expf(Sr[j] - mx); Sr[j] = e; sum += e; }
        sum += __shfl_xor(sum, 1);
        float inv = 1.0f / sum;
        for (int j = 0; j < 64; ++j) Sr[j] *= inv;
    }
    __syncthreads();

    // out = P @ V : wave owns row tile i = wave; A fragment shared by both d tiles.
    {
        int i = wave;
        v8f acc0 = {}, acc1 = {};
#pragma unroll
        for (int kc = 0; kc < 4; ++kc) {
            v16h a  = load_a_f32(S + i * 16 * NTOK + kc * 32, NTOK);
            v16h b0 = load_b_kmajor(vb + kc * 32 * HDP, HDP);
            v16h b1 = load_b_kmajor(vb + kc * 32 * HDP + 16, HDP);
            acc0 = wmma_f16(a, b0, acc0);
            acc1 = wmma_f16(a, b1, acc1);
        }
#pragma unroll
        for (int r = 0; r < 8; ++r) {
            int m = i * 16 + r + ((lane >> 4) << 3);
            int n = lane & 15;
            xout[((long)b * NTOK + m) * C2PAD + CDIM + h * HD + n] = (_Float16)acc0[r];
            int d = 16 + n;
            if (d < HD)
                xout[((long)b * NTOK + m) * C2PAD + CDIM + h * HD + d] = (_Float16)acc1[r];
        }
    }
}

// ---------------- Mutual attention: one block per (window, head, half) ----------------
__global__ void __launch_bounds__(256)
attn_mut_kernel(const _Float16* __restrict__ q, const _Float16* __restrict__ kT,
                const _Float16* __restrict__ v, const float* __restrict__ mask,
                _Float16* __restrict__ xout) {
    __shared__ float S[64 * 64];            // 16 KB
    int id = blockIdx.x;
    int rem = id % (HEADS * 2);
    int b = id / (HEADS * 2);
    int h = rem >> 1;
    int hf = rem & 1;                        // 0 -> x1 (q from top half), 1 -> x2
    long bh = (long)b * HEADS + h;
    int qoff  = hf ? 0  : 64;
    int kvoff = hf ? 64 : 0;
    int ooff  = hf ? 64 : 0;                 // x1 -> tokens 0..63, x2 -> 64..127
    const _Float16* qb = q  + bh * NTOK * HDP + (long)qoff * HDP;
    const _Float16* kb = kT + bh * HDP * NTOK + kvoff;
    const _Float16* vb = v  + bh * NTOK * HDP + (long)kvoff * HDP;
    const float* maskw = mask + (long)(b & (NWIN - 1)) * NTOK * NTOK;  // use [:64,:64]
    int wave = (int)(threadIdx.x >> 5), lane = (int)(threadIdx.x & 31);

    // S tiles: wave -> (col tile j = wave&3, row tiles i0, i0+1); B reused for 2 rows
    {
        int j  = wave & 3;
        int i0 = (wave >> 2) * 2;
        v16h bb = load_b_kmajor(kb + j * 16, NTOK);
#pragma unroll
        for (int ii = 0; ii < 2; ++ii) {
            int i = i0 + ii;
            v16h a = load_a_f16(qb + i * 16 * HDP, HDP);
            v8f acc = {};
            acc = wmma_f16(a, bb, acc);
#pragma unroll
            for (int r = 0; r < 8; ++r) {
                int m = i * 16 + r + ((lane >> 4) << 3);
                int n = j * 16 + (lane & 15);
                S[m * 64 + n] = acc[r] + maskw[m * NTOK + n];
            }
        }
    }
    __syncthreads();

    // softmax: 2 lanes per row over 64 rows (threads 0..127)
    if (threadIdx.x < 128) {
        int row  = (int)(threadIdx.x >> 1);
        int half = (int)(threadIdx.x & 1);
        float* Sr = S + row * 64 + half * 32;
        float mx = -1e30f;
        for (int j = 0; j < 32; ++j) mx = fmaxf(mx, Sr[j]);
        mx = fmaxf(mx, __shfl_xor(mx, 1));
        float sum = 0.0f;
        for (int j = 0; j < 32; ++j) { float e = __expf(Sr[j] - mx); Sr[j] = e; sum += e; }
        sum += __shfl_xor(sum, 1);
        float inv = 1.0f / sum;
        for (int j = 0; j < 32; ++j) Sr[j] *= inv;
    }
    __syncthreads();

    // out = P @ V : 8 tiles, one per wave
    {
        int i = wave >> 1, dj = wave & 1;
        v8f acc = {};
#pragma unroll
        for (int kc = 0; kc < 2; ++kc) {
            v16h a  = load_a_f32(S + i * 16 * 64 + kc * 32, 64);
            v16h bb = load_b_kmajor(vb + kc * 32 * HDP + dj * 16, HDP);
            acc = wmma_f16(a, bb, acc);
        }
#pragma unroll
        for (int r = 0; r < 8; ++r) {
            int m = i * 16 + r + ((lane >> 4) << 3);
            int d = dj * 16 + (lane & 15);
            if (d < HD)
                xout[((long)b * NTOK + ooff + m) * C2PAD + h * HD + d] = (_Float16)acc[r];
        }
    }
}

// ---------------- Projection: out[tok,o] = sum_c Xo[tok,c]*Wp[o,c] + b[o] ----------------
// grid (BWIN, 3); each wave: 1 row tile x 4 col tiles (A reused 4x per K step)
__global__ void __launch_bounds__(256)
proj_gemm_kernel(const _Float16* __restrict__ Xo, const _Float16* __restrict__ W,
                 const float* __restrict__ bproj, float* __restrict__ out) {
    int b    = blockIdx.x;
    int jg   = blockIdx.y;                  // col-tile group 0..2 (12 tiles of 16)
    int wave = (int)(threadIdx.x >> 5);
    int lane = (int)(threadIdx.x & 31);
    int row0 = wave * 16;
    const _Float16* A  = Xo + ((long)b * NTOK + row0) * C2PAD;
    const _Float16* Bw = W + (long)jg * 4 * 16 * C2PAD;
    v8f acc0 = {}, acc1 = {}, acc2 = {}, acc3 = {};
#pragma unroll
    for (int kc = 0; kc < C2PAD / 32; ++kc) {
        v16h a = load_a_f16(A + kc * 32, C2PAD);
        if (kc + 1 < C2PAD / 32)
            __builtin_prefetch(A + (kc + 1) * 32, 0, 3);   // global_prefetch_b8
        v16h b0 = load_b_nmajor(Bw + 0 * 16 * C2PAD + kc * 32, C2PAD);
        v16h b1 = load_b_nmajor(Bw + 1 * 16 * C2PAD + kc * 32, C2PAD);
        v16h b2 = load_b_nmajor(Bw + 2 * 16 * C2PAD + kc * 32, C2PAD);
        v16h b3 = load_b_nmajor(Bw + 3 * 16 * C2PAD + kc * 32, C2PAD);
        acc0 = wmma_f16(a, b0, acc0);
        acc1 = wmma_f16(a, b1, acc1);
        acc2 = wmma_f16(a, b2, acc2);
        acc3 = wmma_f16(a, b3, acc3);
    }
#pragma unroll
    for (int cc = 0; cc < 4; ++cc) {
        v8f acc = (cc == 0) ? acc0 : (cc == 1) ? acc1 : (cc == 2) ? acc2 : acc3;
        int jt = jg * 4 + cc;
#pragma unroll
        for (int r = 0; r < 8; ++r) {
            int m = row0 + r + ((lane >> 4) << 3);
            int o = jt * 16 + (lane & 15);
            if (o < CDIM)
                out[((long)b * NTOK + m) * CDIM + o] = acc[r] + bproj[o];
        }
    }
}

// ---------------- Host launcher ----------------
extern "C" void kernel_launch(void* const* d_in, const int* in_sizes, int n_in,
                              void* d_out, int out_size, void* d_ws, size_t ws_size,
                              hipStream_t stream) {
    const float* x     = (const float*)d_in[0];
    const float* mask  = (const float*)d_in[1];
    const float* rpb   = (const float*)d_in[2];
    const float* Wself = (const float*)d_in[3];
    const float* Wmut  = (const float*)d_in[4];
    const float* Wproj = (const float*)d_in[5];
    const float* bproj = (const float*)d_in[6];
    float* out = (float*)d_out;

    char* ws = (char*)d_ws;
    size_t off = 0;
    auto alloc = [&](size_t bytes) -> void* {
        void* p = ws + off;
        off += (bytes + 255) & ~(size_t)255;
        return p;
    };

    const size_t xbytes   = (size_t)BWIN * NTOK * KPAD * sizeof(_Float16);
    const size_t qkvbytes = (size_t)BWIN * HEADS * NTOK * HDP * sizeof(_Float16);

    _Float16* xf     = (_Float16*)alloc(xbytes);
    _Float16* xmf    = (_Float16*)alloc(xbytes);
    _Float16* WselfF = (_Float16*)alloc((size_t)OQKVP * KPAD * sizeof(_Float16));
    _Float16* WmutF  = (_Float16*)alloc((size_t)OQKVP * KPAD * sizeof(_Float16));
    _Float16* WprojF = (_Float16*)alloc((size_t)192 * C2PAD * sizeof(_Float16));
    float*    biasF  = (float*)alloc((size_t)HEADS * NTOK * NTOK * sizeof(float));
    _Float16* qs  = (_Float16*)alloc(qkvbytes);
    _Float16* ksT = (_Float16*)alloc(qkvbytes);
    _Float16* vs  = (_Float16*)alloc(qkvbytes);
    _Float16* qm  = (_Float16*)alloc(qkvbytes);
    _Float16* kmT = (_Float16*)alloc(qkvbytes);
    _Float16* vm  = (_Float16*)alloc(qkvbytes);
    _Float16* xo  = (_Float16*)alloc((size_t)BWIN * NTOK * C2PAD * sizeof(_Float16));

    const float qscale = 0.18257418583505536f;   // 30^-0.5

    // prep
    {
        long n = (long)BWIN * NTOK * KPAD;
        prep_x_kernel<<<(unsigned)((n + 255) / 256), 256, 0, stream>>>(x, xf, xmf);
    }
    prep_wqkv_kernel<<<(OQKVP * KPAD + 255) / 256, 256, 0, stream>>>(Wself, WselfF);
    prep_wqkv_kernel<<<(OQKVP * KPAD + 255) / 256, 256, 0, stream>>>(Wmut, WmutF);
    prep_wproj_kernel<<<(192 * C2PAD + 255) / 256, 256, 0, stream>>>(Wproj, WprojF);
    prep_bias_kernel<<<(HEADS * NTOK * NTOK + 255) / 256, 256, 0, stream>>>(rpb, biasF);
    pad_qkv_kernel<<<(BWIN * HEADS * NTOK + 255) / 256, 256, 0, stream>>>(qs, ksT, vs);
    pad_qkv_kernel<<<(BWIN * HEADS * NTOK + 255) / 256, 256, 0, stream>>>(qm, kmT, vm);

    // QKV GEMMs (WMMA, 1x4 register blocking, guard-free K loop)
    dim3 gq(BWIN, 9);
    qkv_gemm_kernel<<<gq, 256, 0, stream>>>(xf,  WselfF, qs, ksT, vs, qscale);
    qkv_gemm_kernel<<<gq, 256, 0, stream>>>(xmf, WmutF,  qm, kmT, vm, qscale);

    // Attention (WMMA)
    attn_self_kernel<<<BWIN * HEADS, 256, 0, stream>>>(qs, ksT, vs, biasF, mask, xo);
    attn_mut_kernel<<<BWIN * HEADS * 2, 256, 0, stream>>>(qm, kmT, vm, mask, xo);

    // Projection (WMMA, 1x4 register blocking) + bias
    dim3 gp(BWIN, 3);
    proj_gemm_kernel<<<gp, 256, 0, stream>>>(xo, WprojF, bproj, out);
}
